// Vandermonde_9096740733239
// MI455X (gfx1250) — matrix operand order repaired and checked by reading, not compiled, not run on hardware
//
#include <hip/hip_runtime.h>
#include <cstdint>

// ---------------------------------------------------------------------------
// Vandermonde SSM kernel generation for gfx1250 (MI455X, wave32, WMMA bf16)
//   kernel[h,l] = sum_p W[h,p] * exp(A[p]*dt)^l     (complex)
// Phase 1: build W (256x1024 complex) in bf16, pre-swizzled to WMMA A-fragment
//          layout; three buffers: Re, Im, -Im (negation hoisted out of loop).
// Phase 2: fused Vandermonde-GEMM: generate V tiles with raw hw exp2/sin/cos
//          into LDS in B-fragment layout, multiply with
//          v_wmma_f32_16x16x32_bf16 (4 WMMAs per complex 16x16 tile).
// ---------------------------------------------------------------------------

typedef __attribute__((ext_vector_type(16))) __bf16 v16bf;
typedef __attribute__((ext_vector_type(8)))  float  v8f;

namespace {
constexpr int   kP  = 1024;     // d_state (K of the GEMM)
constexpr int   kH  = 256;      // d_output (M of the GEMM)
constexpr int   kL  = 16384;    // kernel length (N of the GEMM)
constexpr float kDT     = 0.001f;
constexpr float kLog2e  = 1.4426950408889634f;   // 1/ln(2)
constexpr float kInv2Pi = 0.15915494309189535f;  // 1/(2*pi)
}

// pack two f32 -> packed bf16 dword (round-half-up + v_perm_b32 byte select)
__device__ __forceinline__ unsigned pack2bf(float lo, float hi) {
  unsigned ulo = __float_as_uint(lo) + 0x8000u;
  unsigned uhi = __float_as_uint(hi) + 0x8000u;
  return __builtin_amdgcn_perm(uhi, ulo, 0x07060302u);
}

union FragBF {                  // one 16x32 bf16 WMMA operand fragment / lane
  v16bf   v;
  unsigned u[8];
  uint4    q[2];
};

// ---------------------------------------------------------------------------
// Kernel 1: W[h,p] = C[h,p] * ((exp(A*dt)-1)/A) * B[p,h]   (complex)
// Stored as packed-bf16 dword pairs in A-fragment order:
//   dword index = ((mt*32 + kt)*32 + lane)*8 + i
//   h = mt*16 + (lane&15)
//   p = kt*32 + (i>>2)*16 + ((lane>>4)&1)*8 + (i&3)*2 + j   (j = 0,1 packed)
// ---------------------------------------------------------------------------
__global__ __launch_bounds__(256) void build_w_kernel(
    const float* __restrict__ A,   // (P,2)
    const float* __restrict__ B,   // (P,H,2)
    const float* __restrict__ C,   // (H,P,2)
    unsigned* __restrict__ Wr,     // packed bf16 pairs, A-frag layout
    unsigned* __restrict__ Wi,     // +Im
    unsigned* __restrict__ Win) {  // -Im (sign bits pre-flipped)
  const int tid = blockIdx.x * 256 + threadIdx.x;   // 0 .. 131071
  const int i   = tid & 7;
  const int ln  = (tid >> 3) & 31;
  const int kt  = (tid >> 8) & 31;
  const int mt  = tid >> 13;

  const int h     = mt * 16 + (ln & 15);
  const int kbase = kt * 32 + (i >> 2) * 16 + ((ln >> 4) & 1) * 8 + (i & 3) * 2;

  float wr[2], wi[2];
#pragma unroll
  for (int j = 0; j < 2; ++j) {
    const int p  = kbase + j;
    const float ar = A[2 * p + 0];
    const float ai = A[2 * p + 1];
    // A_bar = exp(A*dt)
    const float mg = __expf(ar * kDT);
    float sn, cs;
    __sincosf(ai * kDT, &sn, &cs);
    const float nr  = mg * cs - 1.0f;          // A_bar - 1 (re)
    const float ni  = mg * sn;                 // A_bar - 1 (im)
    const float inv = 1.0f / (ar * ar + ai * ai);
    const float cfr = (nr * ar + ni * ai) * inv;   // (A_bar-1)/A
    const float cfi = (ni * ar - nr * ai) * inv;
    const float br  = B[(p * kH + h) * 2 + 0];
    const float bi  = B[(p * kH + h) * 2 + 1];
    const float b2r = cfr * br - cfi * bi;     // B_bar
    const float b2i = cfr * bi + cfi * br;
    const float cr  = C[(h * kP + p) * 2 + 0];
    const float ci  = C[(h * kP + p) * 2 + 1];
    wr[j] = cr * b2r - ci * b2i;               // W = C * B_bar
    wi[j] = cr * b2i + ci * b2r;
  }
  const unsigned pwi = pack2bf(wi[0], wi[1]);
  Wr[tid]  = pack2bf(wr[0], wr[1]);
  Wi[tid]  = pwi;
  Win[tid] = pwi ^ 0x80008000u;
}

// ---------------------------------------------------------------------------
// Kernel 2: fused Vandermonde GEMM.
// Block = 256 threads (8 wave32), owns 32 L-columns and all 256 H-rows.
// Wave w -> M rows [w*32, w*32+32), both 16-wide N tiles.
// K loop over P in steps of 32 (bf16 WMMA K).
// B-fragment layout (32x16 bf16): lane = 16*(K half) + N col,
//   VGPR i holds K = (i>>2)*16 + (lane>>4)*8 + 2*(i&3), +1 (packed bf16).
// ---------------------------------------------------------------------------
__global__ __launch_bounds__(256) void vand_gemm_kernel(
    const float* __restrict__ A,        // (P,2) -> prescaled lambda
    const unsigned* __restrict__ Wr,    // A-frag layout bf16 pairs
    const unsigned* __restrict__ Wi,
    const unsigned* __restrict__ Win,
    float2* __restrict__ out) {         // (H, L) complex as float2
  __shared__ float2  lamS[kP];          // 8 KB: {ar*dt*log2e, ai*dt/(2pi)}
  __shared__ unsigned VrL[2][32][8];    // 2 KB: V real, B-frag layout
  __shared__ unsigned ViL[2][32][8];    // 2 KB: V imag, B-frag layout

  const int tid    = threadIdx.x;
  const int ln     = tid & 31;
  const int wv     = tid >> 5;               // 0..7
  const int l_base = blockIdx.x * 32;

  const float2* A2 = (const float2*)A;
  for (int idx = tid; idx < kP; idx += 256) {
    const float2 a = A2[idx];
    lamS[idx] = make_float2(a.x * (kDT * kLog2e), a.y * (kDT * kInv2Pi));
  }
  __syncthreads();

  const v8f vzero = {0.f, 0.f, 0.f, 0.f, 0.f, 0.f, 0.f, 0.f};
  v8f accRe[2][2], accIm[2][2];
#pragma unroll
  for (int a = 0; a < 2; ++a)
#pragma unroll
    for (int b = 0; b < 2; ++b) { accRe[a][b] = vzero; accIm[a][b] = vzero; }

  for (int kt = 0; kt < kP / 32; ++kt) {
    const int kc = kt * 32;

    // ---- cooperative V-tile generation: 2 packed dwords / thread ----------
    // V[p,l] = 2^(lx*l) * (cos(2pi*ly*l) + i sin(2pi*ly*l)); |ly*l| <= 8.2
    // revolutions, inside v_sin/cos_f32 native range -> no range reduction.
#pragma unroll
    for (int ss = 0; ss < 2; ++ss) {
      const int s   = tid * 2 + ss;          // 0..511 slot id
      const int nt  = s >> 8;                // N tile
      const int sln = (s >> 3) & 31;         // fragment lane
      const int si  = s & 7;                 // fragment VGPR index
      const int l   = l_base + nt * 16 + (sln & 15);
      const int pb  = kc + (si >> 2) * 16 + ((sln >> 4) & 1) * 8 + (si & 3) * 2;
      const float fl = (float)l;
      float re[2], im[2];
#pragma unroll
      for (int j = 0; j < 2; ++j) {
        const float2 lm = lamS[pb + j];
        const float  mg = __builtin_amdgcn_exp2f(lm.x * fl);
        const float  ph = lm.y * fl;
        re[j] = mg * __builtin_amdgcn_cosf(ph);
        im[j] = mg * __builtin_amdgcn_sinf(ph);
      }
      VrL[nt][sln][si] = pack2bf(re[0], re[1]);
      ViL[nt][sln][si] = pack2bf(im[0], im[1]);
    }
    __syncthreads();

    // ---- load B fragments from LDS ----------------------------------------
    FragBF vr[2], vi[2];
#pragma unroll
    for (int nt = 0; nt < 2; ++nt) {
#pragma unroll
      for (int i = 0; i < 8; ++i) {
        vr[nt].u[i] = VrL[nt][ln][i];
        vi[nt].u[i] = ViL[nt][ln][i];
      }
    }

    // ---- load A fragments (pre-swizzled W) + prefetch next K step ---------
    FragBF wrF[2], wiF[2], wnF[2];
#pragma unroll
    for (int mtl = 0; mtl < 2; ++mtl) {
      const int    mt   = wv * 2 + mtl;
      const size_t base = ((size_t)(mt * 32 + kt) * 32 + ln) * 8;
      const uint4* pr4  = (const uint4*)(Wr + base);
      const uint4* pi4  = (const uint4*)(Wi + base);
      const uint4* pn4  = (const uint4*)(Win + base);
      wrF[mtl].q[0] = pr4[0]; wrF[mtl].q[1] = pr4[1];
      wiF[mtl].q[0] = pi4[0]; wiF[mtl].q[1] = pi4[1];
      wnF[mtl].q[0] = pn4[0]; wnF[mtl].q[1] = pn4[1];
      if (kt + 1 < kP / 32) {
        __builtin_prefetch(Wr  + base + 32 * 8, 0, 3);  // WGP-scope prefetch
        __builtin_prefetch(Wi  + base + 32 * 8, 0, 3);
        __builtin_prefetch(Win + base + 32 * 8, 0, 3);
      }
    }

    // ---- complex MAC: 4 bf16 WMMAs per 16x16 output tile ------------------
    //   accRe += Wr x Vr + (-Wi) x Vi ; accIm += Wr x Vi + Wi x Vr
#pragma unroll
    for (int mtl = 0; mtl < 2; ++mtl) {
#pragma unroll
      for (int nt = 0; nt < 2; ++nt) {
        accRe[mtl][nt] = __builtin_amdgcn_wmma_f32_16x16x32_bf16(
            false, wrF[mtl].v, false, vr[nt].v, (short)0, accRe[mtl][nt],
            false, false);
        accRe[mtl][nt] = __builtin_amdgcn_wmma_f32_16x16x32_bf16(
            false, wnF[mtl].v, false, vi[nt].v, (short)0, accRe[mtl][nt],
            false, false);
        accIm[mtl][nt] = __builtin_amdgcn_wmma_f32_16x16x32_bf16(
            false, wrF[mtl].v, false, vi[nt].v, (short)0, accIm[mtl][nt],
            false, false);
        accIm[mtl][nt] = __builtin_amdgcn_wmma_f32_16x16x32_bf16(
            false, wiF[mtl].v, false, vr[nt].v, (short)0, accIm[mtl][nt],
            false, false);
      }
    }
    __syncthreads();   // protect LDS V tile before next generation pass
  }

  // ---- epilogue: C/D fragment -> interleaved (H, L, 2) float32 ------------
#pragma unroll
  for (int mtl = 0; mtl < 2; ++mtl) {
#pragma unroll
    for (int nt = 0; nt < 2; ++nt) {
      const int l = l_base + nt * 16 + (ln & 15);
#pragma unroll
      for (int v = 0; v < 8; ++v) {
        const int h = wv * 32 + mtl * 16 + v + ((ln >> 4) << 3);
        out[(size_t)h * kL + l] =
            make_float2(accRe[mtl][nt][v], accIm[mtl][nt][v]);
      }
    }
  }
}

// ---------------------------------------------------------------------------
extern "C" void kernel_launch(void* const* d_in, const int* in_sizes, int n_in,
                              void* d_out, int out_size, void* d_ws,
                              size_t ws_size, hipStream_t stream) {
  (void)in_sizes; (void)n_in; (void)out_size; (void)ws_size;
  const float* A = (const float*)d_in[0];   // (P,2)
  const float* B = (const float*)d_in[1];   // (P,H,2)
  const float* C = (const float*)d_in[2];   // (H,P,2)

  unsigned* Wr  = (unsigned*)d_ws;                 // 131072 dwords = 512 KB
  unsigned* Wi  = Wr + (size_t)kH * kP / 2;        // +512 KB
  unsigned* Win = Wi + (size_t)kH * kP / 2;        // +512 KB

  // Phase 1: W in WMMA A-fragment bf16 layout (131072 dwords per part)
  build_w_kernel<<<512, 256, 0, stream>>>(A, B, C, Wr, Wi, Win);

  // Phase 2: fused Vandermonde-GEMM, one block per 32 output columns
  vand_gemm_kernel<<<kL / 32, 256, 0, stream>>>(A, Wr, Wi, Win,
                                                (float2*)d_out);
}
